// Decoder_78305843741218
// MI455X (gfx1250) — matrix operand order repaired and checked by reading, not compiled, not run on hardware
//
#include <hip/hip_runtime.h>
#include <cstdint>
#include <cstddef>

// Problem constants (match reference)
#define H   1024
#define DD  256
#define BSZ 1024
#define TT  256
#define G3H 3072   // 3*H

// GEMM tiling
#define BM 128
#define BN 128
#define KS 64      // K-slice held in LDS per stage (2 wmma-depths)

typedef __bf16 bf16_t;
typedef bf16_t v16bf __attribute__((ext_vector_type(16)));
typedef float  v8f   __attribute__((ext_vector_type(8)));

// ---------------- helpers ----------------

__device__ __forceinline__ unsigned short f2bf(float f) {
  unsigned int u = __float_as_uint(f);
  unsigned int r = u + 0x7FFFu + ((u >> 16) & 1u);   // round-to-nearest-even
  return (unsigned short)(r >> 16);
}

union FragCast {
  struct { uint4 lo; uint4 hi; } q;
  v16bf v;
};

// A-matrix 16x32 bf16 fragment from LDS (row-major, ld = KS).
// ISA 7.12.2: lane half interleaves K by 8:
//   elements 0..7  -> K = ks + half*8 .. +7 ; elements 8..15 -> K = ks + 16 + half*8 .. +7
__device__ __forceinline__ v16bf lds_frag_a(const unsigned short* s, int row, int ks, int half) {
  const unsigned short* p = s + row * KS + ks + half * 8;
  FragCast fc;
  fc.q.lo = *(const uint4*)(p);
  fc.q.hi = *(const uint4*)(p + 16);
  return fc.v;
}

// B-matrix 32x16 bf16 fragment from LDS: lane holds column n=lane&15 (== weight
// row), K interleaved by 8 like A (elements 0..7 -> K = ks + half*8 .., 8..15 -> K = ks + 16 + half*8 ..).
__device__ __forceinline__ v16bf lds_frag_b(const unsigned short* s, int row, int ks, int half) {
  const unsigned short* p = s + row * KS + ks + half * 8;
  FragCast fc;
  fc.q.lo = *(const uint4*)(p);
  fc.q.hi = *(const uint4*)(p + 16);
  return fc.v;
}

// one 8-WMMA sub-block at K-offset ks within the LDS slice
__device__ __forceinline__ void compute_ks(const unsigned short* as, const unsigned short* bs,
                                           int ks, int wm, int wn, int r, int half,
                                           v8f acc[2][4]) {
  v16bf af[2];
#pragma unroll
  for (int mi = 0; mi < 2; ++mi)
    af[mi] = lds_frag_a(as, wm * 32 + mi * 16 + r, ks, half);
#pragma unroll
  for (int ni = 0; ni < 4; ++ni) {
    v16bf bfg = lds_frag_b(bs, wn * 64 + ni * 16 + r, ks, half);
    acc[0][ni] = __builtin_amdgcn_wmma_f32_16x16x32_bf16(
        false, af[0], false, bfg, (short)0, acc[0][ni], false, false);
    acc[1][ni] = __builtin_amdgcn_wmma_f32_16x16x32_bf16(
        false, af[1], false, bfg, (short)0, acc[1][ni], false, false);
  }
}

// ---------------- GEMM: C(MxN) = A(MxK,bf16) @ B(NxK,bf16)^T [+ addM] [+ bias] ----------------
// grid = (N/BN, M/BM), block = 256 (8 waves as 4(m) x 2(n)).
// LDS double-buffered 64-deep K-slices; one barrier per 16 WMMAs. The next
// slice's global fetch is split in two halves interleaved with the two 8-WMMA
// sub-blocks so only 4 uint4 prefetch values are live at any point (no spills).
__global__ __launch_bounds__(256, 1) void gemm_bf16_wmma(
    const unsigned short* __restrict__ A,
    const unsigned short* __restrict__ B,
    float* __restrict__ C,
    const float* __restrict__ addM,   // nullable, MxN
    const float* __restrict__ bias,   // nullable, N
    int N, int K) {
  __shared__ unsigned short As[2][BM * KS];   // 2 x 16KB
  __shared__ unsigned short Bs[2][BN * KS];   // 2 x 16KB

  const int tid  = threadIdx.x;
  const int lane = tid & 31;
  const int wave = tid >> 5;
  const int wm = wave & 3;          // 0..3 -> 32-row m strip
  const int wn = wave >> 2;         // 0..1 -> 64-col n strip
  const int r = lane & 15;
  const int half = lane >> 4;

  const int mBase = blockIdx.y * BM;
  const int nBase = blockIdx.x * BN;

  // Global fetch assignment: tile = BM x KS = 1024 16B-chunks per matrix;
  // 4 chunks/thread. chunk c: row = c>>3, column offset = (c&7)*8 bf16.
  const unsigned short* Ag[4];
  const unsigned short* Bg[4];
  int lofs[4];
#pragma unroll
  for (int j = 0; j < 4; ++j) {
    int c = tid + 256 * j;
    int row = c >> 3;
    int cq = (c & 7) * 8;
    lofs[j] = row * KS + cq;
    Ag[j] = A + (size_t)(mBase + row) * K + cq;
    Bg[j] = B + (size_t)(nBase + row) * K + cq;
  }

  v8f acc[2][4];
#pragma unroll
  for (int mi = 0; mi < 2; ++mi)
#pragma unroll
    for (int ni = 0; ni < 4; ++ni) acc[mi][ni] = v8f{};

  const int nk = K / KS;

  // stage 0 load
  {
#pragma unroll
    for (int j = 0; j < 4; ++j) {
      uint4 av = *(const uint4*)(Ag[j]);
      uint4 bv = *(const uint4*)(Bg[j]);
      *(uint4*)(&As[0][lofs[j]]) = av;
      *(uint4*)(&Bs[0][lofs[j]]) = bv;
    }
  }
  __syncthreads();

  int buf = 0;
  for (int kb = 0; kb < nk; ++kb) {
    const bool more = (kb + 1) < nk;
    const int nb = buf ^ 1;
    const int ko = (kb + 1) * KS;
    const unsigned short* as = As[buf];
    const unsigned short* bs = Bs[buf];

    // half 1 of next-slice fetch (4 x uint4 live)
    uint4 p0, p1, p2, p3;
    if (more) {
      p0 = *(const uint4*)(Ag[0] + ko);
      p1 = *(const uint4*)(Ag[1] + ko);
      p2 = *(const uint4*)(Bg[0] + ko);
      p3 = *(const uint4*)(Bg[1] + ko);
    }

    compute_ks(as, bs, 0, wm, wn, r, half, acc);

    if (more) {
      // LDS[nb] is free: its previous contents were consumed before the last barrier
      *(uint4*)(&As[nb][lofs[0]]) = p0;
      *(uint4*)(&As[nb][lofs[1]]) = p1;
      *(uint4*)(&Bs[nb][lofs[0]]) = p2;
      *(uint4*)(&Bs[nb][lofs[1]]) = p3;
      // half 2 of next-slice fetch
      p0 = *(const uint4*)(Ag[2] + ko);
      p1 = *(const uint4*)(Ag[3] + ko);
      p2 = *(const uint4*)(Bg[2] + ko);
      p3 = *(const uint4*)(Bg[3] + ko);
    }

    compute_ks(as, bs, 32, wm, wn, r, half, acc);

    if (more) {
      *(uint4*)(&As[nb][lofs[2]]) = p0;
      *(uint4*)(&As[nb][lofs[3]]) = p1;
      *(uint4*)(&Bs[nb][lofs[2]]) = p2;
      *(uint4*)(&Bs[nb][lofs[3]]) = p3;
      __syncthreads();
      buf = nb;
    }
  }

  // Epilogue. C layout: VGPR v -> row m = v + 8*half, column n = lane&15.
#pragma unroll
  for (int mi = 0; mi < 2; ++mi) {
#pragma unroll
    for (int ni = 0; ni < 4; ++ni) {
      int n = nBase + wn * 64 + ni * 16 + r;
      float bv = bias ? bias[n] : 0.0f;
#pragma unroll
      for (int v = 0; v < 8; ++v) {
        int m = mBase + wm * 32 + mi * 16 + half * 8 + v;
        float val = acc[mi][ni][v] + bv;
        if (addM) val += addM[(size_t)m * (size_t)N + n];
        C[(size_t)m * (size_t)N + n] = val;
      }
    }
  }
}

// ---------------- GRU elementwise combine ----------------
__global__ __launch_bounds__(256) void gru_combine(
    const float* __restrict__ gi, const float* __restrict__ gh,
    float* __restrict__ h, unsigned short* __restrict__ hbf) {
  int idx = blockIdx.x * 256 + threadIdx.x;       // BSZ*H total
  int b = idx >> 10;
  int j = idx & (H - 1);
  const float* gib = gi + (size_t)b * G3H;
  const float* ghb = gh + (size_t)b * G3H;
  float rg = 1.0f / (1.0f + expf(-(gib[j] + ghb[j])));
  float zg = 1.0f / (1.0f + expf(-(gib[H + j] + ghb[H + j])));
  float ng = tanhf(gib[2 * H + j] + rg * ghb[2 * H + j]);
  float hv = (1.0f - zg) * ng + zg * h[idx];
  h[idx] = hv;
  hbf[idx] = f2bf(hv);
}

// ---------------- output activation: softmax(0:32) | softmax(32:47) | sigmoid(47:256) ----------
__global__ __launch_bounds__(256) void act_kernel(
    const float* __restrict__ fc, float* __restrict__ out,
    unsigned short* __restrict__ xbf, int t) {
  int lane = threadIdx.x & 31;
  int wave = threadIdx.x >> 5;
  int row = blockIdx.x * 8 + wave;
  const float* fr = fc + (size_t)row * DD;
  float* orow = out + ((size_t)row * TT + t) * DD;
  unsigned short* xrow = xbf + (size_t)row * DD;

  // softmax over cols 0..31
  float v = fr[lane];
  float mx = v;
#pragma unroll
  for (int o = 16; o > 0; o >>= 1) mx = fmaxf(mx, __shfl_xor(mx, o, 32));
  float e = expf(v - mx);
  float s = e;
#pragma unroll
  for (int o = 16; o > 0; o >>= 1) s += __shfl_xor(s, o, 32);
  float r0 = e / s;
  orow[lane] = r0;
  xrow[lane] = f2bf(r0);

  // softmax over cols 32..46
  float v2 = (lane < 15) ? fr[32 + lane] : -3.0e38f;
  float mx2 = v2;
#pragma unroll
  for (int o = 16; o > 0; o >>= 1) mx2 = fmaxf(mx2, __shfl_xor(mx2, o, 32));
  float e2 = (lane < 15) ? expf(v2 - mx2) : 0.0f;
  float s2 = e2;
#pragma unroll
  for (int o = 16; o > 0; o >>= 1) s2 += __shfl_xor(s2, o, 32);
  if (lane < 15) {
    float r1 = e2 / s2;
    orow[32 + lane] = r1;
    xrow[32 + lane] = f2bf(r1);
  }

  // sigmoid over cols 47..255
  for (int c = 47 + lane; c < DD; c += 32) {
    float r2 = 1.0f / (1.0f + expf(-fr[c]));
    orow[c] = r2;
    xrow[c] = f2bf(r2);
  }
}

// ---------------- one-time conversion kernels ----------------

__global__ void slice2bf(const float* __restrict__ src, unsigned short* __restrict__ dst,
                         int rows, int cols, int ld, int coff) {
  int idx = blockIdx.x * blockDim.x + threadIdx.x;
  if (idx >= rows * cols) return;
  int rr = idx / cols;
  int cc = idx - rr * cols;
  dst[idx] = f2bf(src[(size_t)rr * (size_t)ld + coff + cc]);
}

__global__ __launch_bounds__(256) void init_embed(
    const float* __restrict__ embed, unsigned short* __restrict__ gbf,
    float* __restrict__ h0f, unsigned short* __restrict__ h0bf,
    float* __restrict__ h1f, unsigned short* __restrict__ h1bf) {
  int idx = blockIdx.x * 256 + threadIdx.x;       // BSZ*H total
  int b = idx >> 10;
  int j = idx & (H - 1);
  const float* e = embed + (size_t)b * (3 * H);
  gbf[idx] = f2bf(e[j]);
  float a0 = e[H + j];     h0f[idx] = a0; h0bf[idx] = f2bf(a0);
  float a1 = e[2 * H + j]; h1f[idx] = a1; h1bf[idx] = f2bf(a1);
}

__global__ __launch_bounds__(256) void init_x(
    const float* __restrict__ dyn, unsigned short* __restrict__ xbf) {
  int idx = blockIdx.x * 256 + threadIdx.x;       // BSZ*DD total
  int b = idx >> 8;
  int d = idx & (DD - 1);
  xbf[idx] = f2bf(dyn[(size_t)b * TT * DD + d]);  // dynamics[b, 0, d]
}

// ---------------- launcher ----------------

extern "C" void kernel_launch(void* const* d_in, const int* in_sizes, int n_in,
                              void* d_out, int out_size, void* d_ws, size_t ws_size,
                              hipStream_t stream) {
  (void)in_sizes; (void)n_in; (void)out_size; (void)ws_size;

  const float* embed    = (const float*)d_in[0];
  const float* dynamics = (const float*)d_in[1];
  // d_in[2] = seq_len (== TT, compile-time)
  const float* W_ih0 = (const float*)d_in[3];
  const float* W_hh0 = (const float*)d_in[4];
  const float* b_ih0 = (const float*)d_in[5];
  const float* b_hh0 = (const float*)d_in[6];
  const float* W_ih1 = (const float*)d_in[7];
  const float* W_hh1 = (const float*)d_in[8];
  const float* b_ih1 = (const float*)d_in[9];
  const float* b_hh1 = (const float*)d_in[10];
  const float* W_fc  = (const float*)d_in[11];
  const float* b_fc  = (const float*)d_in[12];
  float* out = (float*)d_out;

  // workspace bump allocator (256B aligned)
  char* base = (char*)d_ws;
  size_t ofs = 0;
  auto alloc = [&](size_t bytes) -> void* {
    void* p = base + ofs;
    ofs = (ofs + bytes + 255) & ~(size_t)255;
    return p;
  };
  unsigned short* wg0   = (unsigned short*)alloc((size_t)G3H * H * 2);   // W_ih0[:, :H]
  unsigned short* wx0   = (unsigned short*)alloc((size_t)G3H * DD * 2);  // W_ih0[:, H:]
  unsigned short* whh0  = (unsigned short*)alloc((size_t)G3H * H * 2);
  unsigned short* wih1  = (unsigned short*)alloc((size_t)G3H * H * 2);
  unsigned short* whh1  = (unsigned short*)alloc((size_t)G3H * H * 2);
  unsigned short* wfc   = (unsigned short*)alloc((size_t)DD * H * 2);
  unsigned short* globb = (unsigned short*)alloc((size_t)BSZ * H * 2);
  float*          h0f   = (float*)alloc((size_t)BSZ * H * 4);
  unsigned short* h0b   = (unsigned short*)alloc((size_t)BSZ * H * 2);
  float*          h1f   = (float*)alloc((size_t)BSZ * H * 4);
  unsigned short* h1b   = (unsigned short*)alloc((size_t)BSZ * H * 2);
  unsigned short* xb    = (unsigned short*)alloc((size_t)BSZ * DD * 2);
  float*          Ggl   = (float*)alloc((size_t)BSZ * G3H * 4);  // glob@Wg0^T + b_ih0
  float*          gi    = (float*)alloc((size_t)BSZ * G3H * 4);
  float*          gh    = (float*)alloc((size_t)BSZ * G3H * 4);
  float*          fco   = (float*)alloc((size_t)BSZ * DD * 4);

  // ---- one-time prep ----
  slice2bf<<<(G3H * H  + 255) / 256, 256, 0, stream>>>(W_ih0, wg0,  G3H, H,  H + DD, 0);
  slice2bf<<<(G3H * DD + 255) / 256, 256, 0, stream>>>(W_ih0, wx0,  G3H, DD, H + DD, H);
  slice2bf<<<(G3H * H  + 255) / 256, 256, 0, stream>>>(W_hh0, whh0, G3H, H,  H, 0);
  slice2bf<<<(G3H * H  + 255) / 256, 256, 0, stream>>>(W_ih1, wih1, G3H, H,  H, 0);
  slice2bf<<<(G3H * H  + 255) / 256, 256, 0, stream>>>(W_hh1, whh1, G3H, H,  H, 0);
  slice2bf<<<(DD  * H  + 255) / 256, 256, 0, stream>>>(W_fc,  wfc,  DD,  H,  H, 0);
  init_embed<<<(BSZ * H) / 256, 256, 0, stream>>>(embed, globb, h0f, h0b, h1f, h1b);
  init_x<<<(BSZ * DD) / 256, 256, 0, stream>>>(dynamics, xb);

  dim3 blk(256);
  dim3 grid_big(G3H / BN, BSZ / BM);   // 24 x 8
  dim3 grid_fc(DD / BN, BSZ / BM);     //  2 x 8

  // Ggl = glob @ Wg0^T + b_ih0  (hoisted: glob is loop-invariant)
  gemm_bf16_wmma<<<grid_big, blk, 0, stream>>>(globb, wg0, Ggl, nullptr, b_ih0, G3H, H);

  // ---- autoregressive rollout ----
  for (int t = 0; t < TT; ++t) {
    // gi0 = Ggl + x @ Wx0^T   (K=256 only)
    gemm_bf16_wmma<<<grid_big, blk, 0, stream>>>(xb, wx0, gi, Ggl, nullptr, G3H, DD);
    // gh0 = h0 @ W_hh0^T + b_hh0
    gemm_bf16_wmma<<<grid_big, blk, 0, stream>>>(h0b, whh0, gh, nullptr, b_hh0, G3H, H);
    gru_combine<<<(BSZ * H) / 256, 256, 0, stream>>>(gi, gh, h0f, h0b);

    // gi1 = h0n @ W_ih1^T + b_ih1 ; gh1 = h1 @ W_hh1^T + b_hh1
    gemm_bf16_wmma<<<grid_big, blk, 0, stream>>>(h0b, wih1, gi, nullptr, b_ih1, G3H, H);
    gemm_bf16_wmma<<<grid_big, blk, 0, stream>>>(h1b, whh1, gh, nullptr, b_hh1, G3H, H);
    gru_combine<<<(BSZ * H) / 256, 256, 0, stream>>>(gi, gh, h1f, h1b);

    // fc = h1n @ W_fc^T + b_fc ; activation -> out[:, t, :] and next x (bf16)
    gemm_bf16_wmma<<<grid_fc, blk, 0, stream>>>(h1b, wfc, fco, nullptr, b_fc, DD, H);
    act_kernel<<<BSZ / 8, 256, 0, stream>>>(fco, out, xb, t);
  }
}
